// TransformerBlock_87101936763455
// MI455X (gfx1250) — compile-verified, matrix operand order checked by reading, not run on hardware
//
#include <hip/hip_runtime.h>
#include <cstdint>

// ---------------------------------------------------------------------------
// Transformer block for MI455X (gfx1250), wave32, WMMA bf16 path.
// All matmuls (QKV, attention QK^T, PV, proj, FC, FC-proj) run on
// v_wmma_f32_16x16x32_bf16 with f32 accumulation. LayerNorm / softmax /
// GELU stay in f32 VALU (negligible vs ~140 GFLOP of matmul).
// GEMMs use double-buffered LDS fed by CDNA5 async global->LDS copies
// (ASYNCcnt) so tile k+1 streams in while tile k runs on the WMMA pipe.
// ---------------------------------------------------------------------------

typedef __attribute__((ext_vector_type(16))) __bf16 v16bf;
typedef __attribute__((ext_vector_type(8)))  float  v8f;

#define DEV static __device__ __forceinline__

constexpr int Bb    = 2;
constexpr int Tt    = 2048;
constexpr int Cc    = 1024;
constexpr int Hh    = 16;
constexpr int Dd    = 64;
constexpr int Mrows = Bb * Tt;   // 4096
constexpr int TCc   = 3 * Cc;    // 3072

DEV v8f wmma_bf16(v16bf a, v16bf b, v8f c) {
  return __builtin_amdgcn_wmma_f32_16x16x32_bf16(
      /*neg_a=*/false, a, /*neg_b=*/false, b,
      /*c_mod=*/(short)0, c, /*reuse_a=*/false, /*reuse_b=*/false);
}

// 16-bit A/B fragment: lane-half kh holds K = kh*8..+7 (v0..v3) and
// K = 16+kh*8..+7 (v4..v7) -> two 16-byte runs from a K-contiguous row.
DEV v16bf load_frag(const __bf16* base, int kh) {
  union { v16bf v; uint4 q[2]; } u;
  u.q[0] = *reinterpret_cast<const uint4*>(base + kh * 8);
  u.q[1] = *reinterpret_cast<const uint4*>(base + 16 + kh * 8);
  return u.v;
}

DEV __bf16 to_bf16(float f) { return (__bf16)f; }

DEV float gelu_exact(float v) {
  return 0.5f * v * (1.0f + erff(v * 0.70710678118654752f));
}

// ---- CDNA5 async global->LDS copy (ASYNCcnt path) -------------------------
// Per ISA 10.x / async_tensor doc: each EXEC-enabled lane copies 16 B from
// its global address (VADDR pair) to LDS[VDST]. Generic shared pointers on
// AMDGPU carry the LDS byte offset in their low 32 bits.
DEV void async_copy_b128(const void* gsrc, void* ldst) {
  unsigned long long ga = (unsigned long long)reinterpret_cast<uintptr_t>(gsrc);
  unsigned int       la = (unsigned int)reinterpret_cast<uintptr_t>(ldst);
  asm volatile("global_load_async_to_lds_b128 %0, %1, off"
               :: "v"(la), "v"(ga) : "memory");
}
DEV void async_wait0() {
  asm volatile("s_wait_asynccnt 0x0" ::: "memory");
}

// ---------------------------------------------------------------------------
// Weight convert+transpose: fp32 [K][N] -> bf16 [N][K] (K-contiguous B).
// ---------------------------------------------------------------------------
__global__ __launch_bounds__(256) void transpose_f32_bf16_kernel(
    const float* __restrict__ in, unsigned short* __restrict__ outU,
    int K, int N) {
  __bf16* out = reinterpret_cast<__bf16*>(outU);
  __shared__ float tile[32][33];
  const int tx = threadIdx.x & 31;
  const int ty = threadIdx.x >> 5;
  const int n0 = blockIdx.x * 32;
  const int k0 = blockIdx.y * 32;
  #pragma unroll
  for (int i = 0; i < 32; i += 8)
    tile[ty + i][tx] = in[(size_t)(k0 + ty + i) * N + n0 + tx];
  __syncthreads();
  #pragma unroll
  for (int i = 0; i < 32; i += 8)
    out[(size_t)(n0 + ty + i) * K + k0 + tx] = to_bf16(tile[tx][ty + i]);
}

// ---------------------------------------------------------------------------
// LayerNorm: one block per row (C=1024), fp32 in -> bf16 out.
// ---------------------------------------------------------------------------
__global__ __launch_bounds__(256) void layernorm_bf16_kernel(
    const float* __restrict__ x, const float* __restrict__ g,
    const float* __restrict__ beta, unsigned short* __restrict__ outU) {
  __bf16* out = reinterpret_cast<__bf16*>(outU);
  const int row = blockIdx.x;
  const float* xr = x + (size_t)row * Cc;
  __shared__ float sh[512];
  float s = 0.f, sq = 0.f;
  for (int c = threadIdx.x; c < Cc; c += 256) {
    float v = xr[c];
    s += v; sq += v * v;
  }
  sh[threadIdx.x] = s;
  sh[256 + threadIdx.x] = sq;
  __syncthreads();
  for (int k = 128; k > 0; k >>= 1) {
    if (threadIdx.x < k) {
      sh[threadIdx.x] += sh[threadIdx.x + k];
      sh[256 + threadIdx.x] += sh[256 + threadIdx.x + k];
    }
    __syncthreads();
  }
  const float mu   = sh[0] * (1.0f / Cc);
  const float var  = sh[256] * (1.0f / Cc) - mu * mu;
  const float rstd = rsqrtf(var + 1e-5f);
  for (int c = threadIdx.x; c < Cc; c += 256)
    out[(size_t)row * Cc + c] = to_bf16((xr[c] - mu) * rstd * g[c] + beta[c]);
}

// ---------------------------------------------------------------------------
// WMMA GEMM: out[M][N] = A[M][K](bf16) * Bt[N][K]^T(bf16) + bias, epilogues:
//   EPI 0: bias              -> bf16 out   (QKV)
//   EPI 1: bias + GELU       -> bf16 out   (FC)
//   EPI 2: bias + residual   -> f32  out   (proj, FC-proj)
// Block tile 128x128, 8 waves of 32(M)x64(N) each -> 2x4 C fragments.
// Double-buffered LDS (2 x 12 KB x 2 = 48 KB) fed by async global->LDS
// b128 copies; compute on buf while buf^1 streams in.
// ---------------------------------------------------------------------------
template <int EPI>
__global__ __launch_bounds__(256) void gemm_wmma_bf16_kernel(
    const unsigned short* __restrict__ Au,
    const unsigned short* __restrict__ Btu,
    const float* __restrict__ bias,
    const float* __restrict__ res,
    float* __restrict__ outF,
    unsigned short* __restrict__ outHu,
    int M, int N, int K) {
  const __bf16* A  = reinterpret_cast<const __bf16*>(Au);
  const __bf16* Bt = reinterpret_cast<const __bf16*>(Btu);
  __bf16* outH = reinterpret_cast<__bf16*>(outHu);

  constexpr int BM = 128, BN = 128, BK = 32, LD = 48;
  __shared__ __bf16 sA[2][BM * LD];
  __shared__ __bf16 sB[2][BN * LD];

  const int tid  = threadIdx.x;
  const int wid  = tid >> 5;
  const int lane = tid & 31;
  const int ln   = lane & 15;
  const int kh   = lane >> 4;
  const int m0 = blockIdx.y * BM;
  const int n0 = blockIdx.x * BN;
  const int wm = (wid & 3) * 32;   // 4 waves along M
  const int wn = (wid >> 2) * 64;  // 2 waves along N

  // this thread's 2 chunks per matrix per tile: u in {tid, tid+256}
  const int r0 = tid >> 2, c0 = (tid & 3) * 8;          // rows 0..63
  const int r1 = (tid + 256) >> 2, c1 = ((tid + 256) & 3) * 8;  // rows 64..127

  auto stage = [&](int buf, int k0) {
    async_copy_b128(&A[(size_t)(m0 + r0) * K + k0 + c0], &sA[buf][r0 * LD + c0]);
    async_copy_b128(&A[(size_t)(m0 + r1) * K + k0 + c1], &sA[buf][r1 * LD + c1]);
    async_copy_b128(&Bt[(size_t)(n0 + r0) * K + k0 + c0], &sB[buf][r0 * LD + c0]);
    async_copy_b128(&Bt[(size_t)(n0 + r1) * K + k0 + c1], &sB[buf][r1 * LD + c1]);
  };

  const v8f vz = {0.f, 0.f, 0.f, 0.f, 0.f, 0.f, 0.f, 0.f};
  v8f acc[2][4];
  #pragma unroll
  for (int i = 0; i < 2; ++i)
    #pragma unroll
    for (int j = 0; j < 4; ++j) acc[i][j] = vz;

  stage(0, 0);
  async_wait0();
  __syncthreads();

  int buf = 0;
  for (int k0 = 0; k0 < K; k0 += BK) {
    if (k0 + BK < K) stage(buf ^ 1, k0 + BK);          // overlap next tile
    if (k0 + 2 * BK < K) {                              // L2 prefetch of k+2
      __builtin_prefetch(&A[(size_t)(m0 + r0) * K + k0 + 2 * BK + c0], 0, 1);
      __builtin_prefetch(&Bt[(size_t)(n0 + r0) * K + k0 + 2 * BK + c0], 0, 1);
    }

    v16bf af[2], bf[4];
    #pragma unroll
    for (int i = 0; i < 2; ++i)
      af[i] = load_frag(&sA[buf][(wm + i * 16 + ln) * LD], kh);
    #pragma unroll
    for (int j = 0; j < 4; ++j)
      bf[j] = load_frag(&sB[buf][(wn + j * 16 + ln) * LD], kh);

    #pragma unroll
    for (int i = 0; i < 2; ++i)
      #pragma unroll
      for (int j = 0; j < 4; ++j)
        acc[i][j] = wmma_bf16(af[i], bf[j], acc[i][j]);

    async_wait0();       // our async copies for buf^1 are done
    __syncthreads();     // everyone finished reading buf and writing buf^1
    buf ^= 1;
  }

  // Epilogue. C layout: lane half kh, vgpr e -> row m = kh*8+e, col n = ln.
  #pragma unroll
  for (int i = 0; i < 2; ++i) {
    #pragma unroll
    for (int j = 0; j < 4; ++j) {
      const int col = n0 + wn + j * 16 + ln;
      const float bv = bias[col];
      #pragma unroll
      for (int e = 0; e < 8; ++e) {
        const int row = m0 + wm + i * 16 + kh * 8 + e;
        float v = acc[i][j][e] + bv;
        if constexpr (EPI == 1) v = gelu_exact(v);
        if constexpr (EPI == 2) {
          v += res[(size_t)row * N + col];
          outF[(size_t)row * N + col] = v;
        } else {
          outH[(size_t)row * N + col] = to_bf16(v);
        }
      }
    }
  }
}

// ---------------------------------------------------------------------------
// Flash attention (causal), bf16 QKV in, bf16 Y out.
// Block = (b, h, 128 q rows); 8 waves, each owns 16 q rows.
// K tile staged via async global->LDS; S = Q K^T (2 WMMAs / 16-key subtile),
// online softmax with half-wave shfl_xor row reductions, P through per-wave
// LDS (C-layout -> A-layout), O += P V via 4 WMMAs. Fully-unmasked key tiles
// skip the causal cndmask block (wave-uniform branch).
// ---------------------------------------------------------------------------
__global__ __launch_bounds__(256) void flash_attn_kernel(
    const unsigned short* __restrict__ qkvU, unsigned short* __restrict__ yU) {
  const __bf16* qkv = reinterpret_cast<const __bf16*>(qkvU);
  __bf16* y = reinterpret_cast<__bf16*>(yU);

  constexpr int LDK = 72;  // 144 B rows
  constexpr int LDV = 48;  // 96 B rows
  constexpr int LDP = 48;
  __shared__ __bf16 sK[32 * LDK];      // [key][d]
  __shared__ __bf16 sV[Dd * LDV];      // transposed: [d][key]
  __shared__ __bf16 sP[8][16 * LDP];   // per-wave P scratch [qrow][key]

  const int tid  = threadIdx.x;
  const int wid  = tid >> 5;
  const int lane = tid & 31;
  const int ln   = lane & 15;
  const int kh   = lane >> 4;
  const int kh8  = kh * 8;

  const int qt = blockIdx.x * 128;
  const int bh = blockIdx.y;
  const int b  = bh >> 4;
  const int h  = bh & 15;
  const size_t rowbase = (size_t)b * Tt;

  const int qbase_w = qt + wid * 16;
  const __bf16* qp = qkv + (rowbase + qbase_w + ln) * TCc + h * Dd;
  v16bf qa0 = load_frag(qp, kh);        // d 0..31
  v16bf qa1 = load_frag(qp + 32, kh);   // d 32..63

  const v8f vz = {0.f, 0.f, 0.f, 0.f, 0.f, 0.f, 0.f, 0.f};
  v8f o[4] = {vz, vz, vz, vz};
  float mi[8], li[8];
  #pragma unroll
  for (int i = 0; i < 8; ++i) { mi[i] = -1e30f; li[i] = 0.f; }

  // this thread's K-tile chunk: 256 x 16 B
  const int kr = tid >> 3, kc = (tid & 7) * 8;

  const int nkt = (qt + 128) >> 5;
  for (int kt = 0; kt < nkt; ++kt) {
    const int kb = kt * 32;
    // stage K tile [32][64] via async global->LDS
    async_copy_b128(&qkv[(rowbase + kb + kr) * TCc + Cc + h * Dd + kc],
                    &sK[kr * LDK + kc]);
    // stage V transposed [d][key] (scalar; needs layout change)
    for (int e = tid; e < 32 * Dd; e += 256) {
      const int key = e >> 6;
      const int d   = e & 63;
      sV[d * LDV + key] = qkv[(rowbase + kb + key) * TCc + 2 * Cc + h * Dd + d];
    }
    async_wait0();
    __syncthreads();

    // ---- S = Q K^T for both 16-key subtiles ----
    v8f st[2];
    const bool need_mask = (kb + 31) > qbase_w;  // any key above the diagonal?
    #pragma unroll
    for (int nf = 0; nf < 2; ++nf) {
      v8f s = vz;
      const __bf16* krow = &sK[(nf * 16 + ln) * LDK];
      s = wmma_bf16(qa0, load_frag(krow, kh), s);
      s = wmma_bf16(qa1, load_frag(krow + 32, kh), s);
      const int key = kb + nf * 16 + ln;
      #pragma unroll
      for (int i = 0; i < 8; ++i) s[i] = s[i] * 0.125f;  // 1/sqrt(64)
      if (need_mask) {
        #pragma unroll
        for (int i = 0; i < 8; ++i)
          if (key > qbase_w + kh8 + i) s[i] = -1e30f;
      }
      st[nf] = s;
    }

    // ---- one combined online-softmax update over the 32-key tile ----
    #pragma unroll
    for (int i = 0; i < 8; ++i) {
      float mx = fmaxf(st[0][i], st[1][i]);
      #pragma unroll
      for (int d = 1; d < 16; d <<= 1) mx = fmaxf(mx, __shfl_xor(mx, d, 32));
      const float mnew  = fmaxf(mi[i], mx);
      const float alpha = __expf(mi[i] - mnew);
      const float p0 = __expf(st[0][i] - mnew);
      const float p1 = __expf(st[1][i] - mnew);
      float rs = p0 + p1;
      #pragma unroll
      for (int d = 1; d < 16; d <<= 1) rs += __shfl_xor(rs, d, 32);
      li[i] = li[i] * alpha + rs;
      mi[i] = mnew;
      #pragma unroll
      for (int df = 0; df < 4; ++df) o[df][i] *= alpha;
      sP[wid][(kh8 + i) * LDP + ln]      = to_bf16(p0);
      sP[wid][(kh8 + i) * LDP + 16 + ln] = to_bf16(p1);
    }

    // ---- O += P V (same-wave LDS ops stay in order) ----
    v16bf pa = load_frag(&sP[wid][ln * LDP], kh);
    #pragma unroll
    for (int df = 0; df < 4; ++df) {
      v16bf vf = load_frag(&sV[(df * 16 + ln) * LDV], kh);
      o[df] = wmma_bf16(pa, vf, o[df]);
    }
    __syncthreads();
  }

  #pragma unroll
  for (int df = 0; df < 4; ++df) {
    #pragma unroll
    for (int i = 0; i < 8; ++i) {
      const float v = o[df][i] / li[i];
      y[(rowbase + qbase_w + kh8 + i) * Cc + h * Dd + df * 16 + ln] =
          to_bf16(v);
    }
  }
}

// ---------------------------------------------------------------------------
// Host-side orchestration
// ---------------------------------------------------------------------------
extern "C" void kernel_launch(void* const* d_in, const int* in_sizes, int n_in,
                              void* d_out, int out_size, void* d_ws,
                              size_t ws_size, hipStream_t stream) {
  (void)in_sizes; (void)n_in; (void)out_size; (void)ws_size;
  const float* x      = (const float*)d_in[0];
  const float* w_attn = (const float*)d_in[1];
  const float* b_attn = (const float*)d_in[2];
  const float* w_proj = (const float*)d_in[3];
  const float* b_proj = (const float*)d_in[4];
  const float* w_fc   = (const float*)d_in[5];
  const float* b_fc   = (const float*)d_in[6];
  const float* w_fcp  = (const float*)d_in[7];
  const float* b_fcp  = (const float*)d_in[8];
  const float* g1     = (const float*)d_in[9];
  const float* be1    = (const float*)d_in[10];
  const float* g2     = (const float*)d_in[11];
  const float* be2    = (const float*)d_in[12];
  float* out = (float*)d_out;

  char* ws = (char*)d_ws;
  size_t off = 0;
  auto alloc = [&](size_t bytes) -> char* {
    char* p = ws + off;
    off += (bytes + 255) & ~(size_t)255;
    return p;
  };
  unsigned short* wAT  = (unsigned short*)alloc((size_t)TCc * Cc * 2);
  unsigned short* wPT  = (unsigned short*)alloc((size_t)Cc * Cc * 2);
  unsigned short* wFT  = (unsigned short*)alloc((size_t)4 * Cc * Cc * 2);
  unsigned short* wFPT = (unsigned short*)alloc((size_t)Cc * 4 * Cc * 2);
  unsigned short* hL   = (unsigned short*)alloc((size_t)Mrows * Cc * 2);
  unsigned short* qkvH = (unsigned short*)alloc((size_t)Mrows * TCc * 2);
  unsigned short* yH   = (unsigned short*)alloc((size_t)Mrows * Cc * 2);
  float*          x1   = (float*)alloc((size_t)Mrows * Cc * 4);
  unsigned short* h2   = (unsigned short*)alloc((size_t)Mrows * Cc * 2);
  unsigned short* h3   = (unsigned short*)alloc((size_t)Mrows * 4 * Cc * 2);

  const dim3 blk(256);

  transpose_f32_bf16_kernel<<<dim3(TCc / 32, Cc / 32), blk, 0, stream>>>(
      w_attn, wAT, Cc, TCc);
  transpose_f32_bf16_kernel<<<dim3(Cc / 32, Cc / 32), blk, 0, stream>>>(
      w_proj, wPT, Cc, Cc);
  transpose_f32_bf16_kernel<<<dim3(4 * Cc / 32, Cc / 32), blk, 0, stream>>>(
      w_fc, wFT, Cc, 4 * Cc);
  transpose_f32_bf16_kernel<<<dim3(Cc / 32, 4 * Cc / 32), blk, 0, stream>>>(
      w_fcp, wFPT, 4 * Cc, Cc);

  layernorm_bf16_kernel<<<dim3(Mrows), blk, 0, stream>>>(x, g1, be1, hL);

  gemm_wmma_bf16_kernel<0><<<dim3(TCc / 128, Mrows / 128), blk, 0, stream>>>(
      hL, wAT, b_attn, nullptr, nullptr, qkvH, Mrows, TCc, Cc);

  flash_attn_kernel<<<dim3(Tt / 128, Bb * Hh), blk, 0, stream>>>(qkvH, yH);

  gemm_wmma_bf16_kernel<2><<<dim3(Cc / 128, Mrows / 128), blk, 0, stream>>>(
      yH, wPT, b_proj, x, x1, nullptr, Mrows, Cc, Cc);

  layernorm_bf16_kernel<<<dim3(Mrows), blk, 0, stream>>>(x1, g2, be2, h2);

  gemm_wmma_bf16_kernel<1><<<dim3(4 * Cc / 128, Mrows / 128), blk, 0, stream>>>(
      h2, wFT, b_fc, nullptr, nullptr, h3, Mrows, 4 * Cc, Cc);

  gemm_wmma_bf16_kernel<2><<<dim3(Cc / 128, Mrows / 128), blk, 0, stream>>>(
      h3, wFPT, b_fcp, x1, out, nullptr, Mrows, Cc, 4 * Cc);
}
